// HeteroGraphTransformer_33303176413375
// MI455X (gfx1250) — matrix operand order repaired and checked by reading, not compiled, not run on hardware
//
#include <hip/hip_runtime.h>
#include <hip/hip_bf16.h>
#include <math.h>

// ---------------------------------------------------------------------------
// HGT forward for MI455X (gfx1250, wave32, WMMA).
// Dense GEMMs: v_wmma_f32_16x16x32_f16, f32->f16 conversion packed into
// ds_store_b128 during LDS staging, f32 accumulate.
// Edge attention: wave-per-edge gathers + f32 atomics (segment softmax).
// ---------------------------------------------------------------------------

typedef __attribute__((ext_vector_type(16))) _Float16 v16h;
typedef __attribute__((ext_vector_type(8)))  _Float16 v8h;
typedef __attribute__((ext_vector_type(8)))  float    v8f;

#define ND   20000   // drugs
#define NP   25000   // proteins
#define NE   150000  // edges per relation
#define NLBL 150000  // label pairs
#define HID  256
#define NHEAD 4
#define DHEAD 64

__device__ __forceinline__ float fast_rcp(float x) { return __builtin_amdgcn_rcpf(x); }

// branch-free tanh-approx gelu: tanh(y) = 1 - 2/(exp(2y)+1), rcp via v_rcp_f32
__device__ __forceinline__ float gelu_tanh(float x) {
  const float y = 0.7978845608028654f * (x + 0.044715f * x * x * x);
  const float t = __expf(2.0f * y);
  const float th = 1.0f - 2.0f * fast_rcp(t + 1.0f);
  return 0.5f * x * (1.0f + th);
}

union FragH { v16h v; v8h h[2]; };

// ---------------------------------------------------------------------------
// Generic C[M,N] = (gelu?)(A[M,K]) @ W[K,N] + bias, optional sigmoid-skip mix:
//   C = beta * C + (1-beta) * X,  beta = sigmoid(*skipPtr)
// Block: 256 threads (8 waves). Tile 64x64, K-step 32.
// Requires: K % 32 == 0, N % 64 == 0. M guarded.
// ---------------------------------------------------------------------------
__global__ __launch_bounds__(256)
void wmma_gemm_bias(const float* __restrict__ A, const float* __restrict__ Wg,
                    const float* __restrict__ bias, float* __restrict__ C,
                    int M, int K, int N, int applyGelu,
                    const float* __restrict__ mixX, const float* __restrict__ skipPtr) {
  __shared__ __align__(16) _Float16 lA[64][32];   // row-major A tile
  __shared__ __align__(16) _Float16 lBt[64][32];  // W tile TRANSPOSED: lBt[n][k]

  const int tid  = threadIdx.x;
  const int lane = tid & 31;
  const int wave = tid >> 5;
  const int mBase = blockIdx.y * 64;
  const int nBase = blockIdx.x * 64;

  const int ct  = wave & 3;          // column 16-tile 0..3
  const int rt0 = (wave >> 2) * 2;   // row 16-tiles rt0, rt0+1

  v8f acc0 = {}; v8f acc1 = {};

  // A staging: thread -> (row, 8 consecutive k)  => one ds_store_b128
  const int sa_m = tid >> 2;          // 0..63
  const int sa_k = (tid & 3) * 8;     // 0,8,16,24
  // W staging: thread -> (one col n, 8 consecutive k) => one ds_store_b128;
  // global reads coalesced across lanes for each k row.
  const int sb_n = tid & 63;          // 0..63
  const int sb_k = (tid >> 6) * 8;    // 0,8,16,24

  for (int k0 = 0; k0 < K; k0 += 32) {
    __syncthreads();
    // stage A (f32 -> f16, optional gelu), packed 16B store.
    // Loads first (vectorizable), gelu under one uniform branch.
    {
      const int gm = mBase + sa_m;
      float vals[8];
      if (gm < M) {
        const float* src = A + (size_t)gm * K + k0 + sa_k;
#pragma unroll
        for (int j = 0; j < 8; ++j) vals[j] = src[j];
        if (applyGelu) {
#pragma unroll
          for (int j = 0; j < 8; ++j) vals[j] = gelu_tanh(vals[j]);
        }
      } else {
#pragma unroll
        for (int j = 0; j < 8; ++j) vals[j] = 0.0f;
      }
      v8h pk;
#pragma unroll
      for (int j = 0; j < 8; ++j) pk[j] = (_Float16)vals[j];
      *(v8h*)&lA[sa_m][sa_k] = pk;
    }
    // stage W transposed, packed 16B store per thread
    {
      const float* src = Wg + (size_t)(k0 + sb_k) * N + nBase + sb_n;
      v8h pk;
#pragma unroll
      for (int j = 0; j < 8; ++j) pk[j] = (_Float16)src[(size_t)j * N];
      *(v8h*)&lBt[sb_n][sb_k] = pk;
    }
    __syncthreads();

    // build fragments per ISA 7.12.2 (16-bit A 16x32; B 32x16)
    const int mrow  = lane & 15;
    const int khalf = lane >> 4;
    FragH aF0, aF1, bF;
    aF0.h[0] = *(const v8h*)&lA[rt0 * 16 + mrow][khalf * 8];
    aF0.h[1] = *(const v8h*)&lA[rt0 * 16 + mrow][16 + khalf * 8];
    aF1.h[0] = *(const v8h*)&lA[(rt0 + 1) * 16 + mrow][khalf * 8];
    aF1.h[1] = *(const v8h*)&lA[(rt0 + 1) * 16 + mrow][16 + khalf * 8];
    bF.h[0]  = *(const v8h*)&lBt[ct * 16 + mrow][khalf * 8];
    bF.h[1]  = *(const v8h*)&lBt[ct * 16 + mrow][16 + khalf * 8];

    acc0 = __builtin_amdgcn_wmma_f32_16x16x32_f16(false, aF0.v, false, bF.v,
                                                  (short)0, acc0, false, false);
    acc1 = __builtin_amdgcn_wmma_f32_16x16x32_f16(false, aF1.v, false, bF.v,
                                                  (short)0, acc1, false, false);
  }

  // epilogue
  float beta = 1.0f, omb = 0.0f;
  if (mixX) {
    float s = *skipPtr;
    beta = fast_rcp(1.0f + __expf(-s));
    omb = 1.0f - beta;
  }
  const int khalf = lane >> 4;
  const int col   = nBase + ct * 16 + (lane & 15);
  const float bv  = bias ? bias[col] : 0.0f;
#pragma unroll
  for (int r = 0; r < 8; ++r) {
    const int row0 = mBase + rt0 * 16 + r + 8 * khalf;
    if (row0 < M) {
      float v = acc0[r] + bv;
      if (mixX) v = beta * v + omb * mixX[(size_t)row0 * N + col];
      C[(size_t)row0 * N + col] = v;
    }
    const int row1 = row0 + 16;
    if (row1 < M) {
      float v = acc1[r] + bv;
      if (mixX) v = beta * v + omb * mixX[(size_t)row1 * N + col];
      C[(size_t)row1 * N + col] = v;
    }
  }
}

// ---------------------------------------------------------------------------
// Per-head relation transform: Y[n, h*64+e] = sum_d X[n, h*64+d] * rel[h][d][e]
// grid = (NHEAD, ceil(M/64)). Full K=64 staged before compute -> in-place safe
// (each block owns a disjoint 64-row x one-head region).
// ---------------------------------------------------------------------------
__global__ __launch_bounds__(256)
void wmma_headwise(const float* __restrict__ X, const float* __restrict__ rel,
                   float* __restrict__ Y, int M) {
  __shared__ __align__(16) _Float16 lA[64][64];
  __shared__ __align__(16) _Float16 lBt[64][64];  // rel transposed: lBt[e][d]

  const int tid  = threadIdx.x;
  const int lane = tid & 31;
  const int wave = tid >> 5;
  const int h     = blockIdx.x;
  const int mBase = blockIdx.y * 64;
  const float* relh = rel + (size_t)h * DHEAD * DHEAD;

  // stage X tile: thread -> (row, 16 consecutive cols) => two ds_store_b128
  {
    const int srow = tid >> 2;          // 0..63
    const int scq  = (tid & 3) * 16;    // 0,16,32,48
    const int gm = mBase + srow;
    v8h p0, p1;
    if (gm < M) {
      const float* src = X + (size_t)gm * HID + h * DHEAD + scq;
#pragma unroll
      for (int j = 0; j < 8; ++j) { p0[j] = (_Float16)src[j]; p1[j] = (_Float16)src[8 + j]; }
    } else {
#pragma unroll
      for (int j = 0; j < 8; ++j) { p0[j] = (_Float16)0.0f; p1[j] = (_Float16)0.0f; }
    }
    *(v8h*)&lA[srow][scq]     = p0;
    *(v8h*)&lA[srow][scq + 8] = p1;
  }
  // stage rel transposed: thread -> (one col e, 16 consecutive d);
  // global reads coalesced across lanes per d row; two packed 16B stores.
  {
    const int se  = tid & 63;           // 0..63
    const int sdq = (tid >> 6) * 16;    // 0,16,32,48
    const float* src = relh + (size_t)sdq * DHEAD + se;
    v8h p0, p1;
#pragma unroll
    for (int j = 0; j < 8; ++j) {
      p0[j] = (_Float16)src[(size_t)j * DHEAD];
      p1[j] = (_Float16)src[(size_t)(8 + j) * DHEAD];
    }
    *(v8h*)&lBt[se][sdq]     = p0;
    *(v8h*)&lBt[se][sdq + 8] = p1;
  }
  __syncthreads();

  const int ct  = wave & 3;
  const int rt0 = (wave >> 2) * 2;
  v8f acc0 = {}; v8f acc1 = {};
  const int mrow  = lane & 15;
  const int khalf = lane >> 4;
#pragma unroll
  for (int k0 = 0; k0 < 64; k0 += 32) {
    FragH aF0, aF1, bF;
    aF0.h[0] = *(const v8h*)&lA[rt0 * 16 + mrow][k0 + khalf * 8];
    aF0.h[1] = *(const v8h*)&lA[rt0 * 16 + mrow][k0 + 16 + khalf * 8];
    aF1.h[0] = *(const v8h*)&lA[(rt0 + 1) * 16 + mrow][k0 + khalf * 8];
    aF1.h[1] = *(const v8h*)&lA[(rt0 + 1) * 16 + mrow][k0 + 16 + khalf * 8];
    bF.h[0]  = *(const v8h*)&lBt[ct * 16 + mrow][k0 + khalf * 8];
    bF.h[1]  = *(const v8h*)&lBt[ct * 16 + mrow][k0 + 16 + khalf * 8];
    acc0 = __builtin_amdgcn_wmma_f32_16x16x32_f16(false, aF0.v, false, bF.v,
                                                  (short)0, acc0, false, false);
    acc1 = __builtin_amdgcn_wmma_f32_16x16x32_f16(false, aF1.v, false, bF.v,
                                                  (short)0, acc1, false, false);
  }

  const int col = h * DHEAD + ct * 16 + (lane & 15);
#pragma unroll
  for (int r = 0; r < 8; ++r) {
    const int row0 = mBase + rt0 * 16 + r + 8 * khalf;
    if (row0 < M) Y[(size_t)row0 * HID + col] = acc0[r];
    const int row1 = row0 + 16;
    if (row1 < M) Y[(size_t)row1 * HID + col] = acc1[r];
  }
}

// ---------------------------------------------------------------------------
// Edge attention logits: alpha[e,h] = <krel[src[e],h,:], q[dst[e],h,:]> * prel[h]/8
// One wave32 per edge; float2 lane loads; shfl_xor reduction.
// ---------------------------------------------------------------------------
__global__ __launch_bounds__(256)
void edge_logits_kernel(const float* __restrict__ krel, const float* __restrict__ q,
                        const int* __restrict__ src, const int* __restrict__ dst,
                        const float* __restrict__ prel, float* __restrict__ alpha, int E) {
  const int gw   = (blockIdx.x * blockDim.x + threadIdx.x) >> 5;
  const int lane = threadIdx.x & 31;
  if (gw >= E) return;
  const int s = src[gw], d = dst[gw];
  const float* kr = krel + (size_t)s * HID;
  const float* qr = q    + (size_t)d * HID;
#pragma unroll
  for (int h = 0; h < NHEAD; ++h) {
    const float2 a = *(const float2*)(kr + h * DHEAD + lane * 2);
    const float2 b = *(const float2*)(qr + h * DHEAD + lane * 2);
    float p = a.x * b.x + a.y * b.y;
#pragma unroll
    for (int m = 16; m > 0; m >>= 1) p += __shfl_xor(p, m, 32);
    if (lane == 0) alpha[(size_t)gw * NHEAD + h] = p * prel[h] * 0.125f;
  }
}

__device__ __forceinline__ void atomicMaxFloat(float* addr, float val) {
  if (val >= 0.0f) atomicMax((int*)addr, __float_as_int(val));
  else             atomicMin((unsigned int*)addr, __float_as_uint(val));
}

__global__ void seg_max_kernel(const float* __restrict__ alpha, const int* __restrict__ dst,
                               float* __restrict__ segmax, int E) {
  const int t = blockIdx.x * blockDim.x + threadIdx.x;
  if (t >= E * NHEAD) return;
  const int e = t >> 2, h = t & 3;
  atomicMaxFloat(&segmax[(size_t)dst[e] * NHEAD + h], alpha[t]);
}

__global__ void seg_exp_sum_kernel(float* __restrict__ alpha, const int* __restrict__ dst,
                                   const float* __restrict__ segmax, float* __restrict__ segsum,
                                   int E) {
  const int t = blockIdx.x * blockDim.x + threadIdx.x;
  if (t >= E * NHEAD) return;
  const int e = t >> 2, h = t & 3;
  const float m = segmax[(size_t)dst[e] * NHEAD + h];  // finite for any non-empty segment
  const float ex = __expf(alpha[t] - m);
  alpha[t] = ex;
  atomicAdd(&segsum[(size_t)dst[e] * NHEAD + h], ex);
}

// msg[dst,h,:] += (ex/segsum) * vrel[src,h,:]   (wave per edge, float2 lanes)
__global__ __launch_bounds__(256)
void seg_aggregate_kernel(const float* __restrict__ alpha, const float* __restrict__ segsum,
                          const float* __restrict__ vrel, const int* __restrict__ src,
                          const int* __restrict__ dst, float* __restrict__ msg, int E) {
  const int gw   = (blockIdx.x * blockDim.x + threadIdx.x) >> 5;
  const int lane = threadIdx.x & 31;
  if (gw >= E) return;
  const int s = src[gw], d = dst[gw];
#pragma unroll
  for (int h = 0; h < NHEAD; ++h) {
    const float w = alpha[(size_t)gw * NHEAD + h] *
                    fast_rcp(segsum[(size_t)d * NHEAD + h] + 1e-16f);
    const float2 v = *(const float2*)(vrel + (size_t)s * HID + h * DHEAD + lane * 2);
    float* mp = msg + (size_t)d * HID + h * DHEAD + lane * 2;
    atomicAdd(mp,     w * v.x);
    atomicAdd(mp + 1, w * v.y);
  }
}

__global__ void fill_kernel(float* __restrict__ p, float v, int n) {
  const int i = blockIdx.x * blockDim.x + threadIdx.x;
  if (i < n) p[i] = v;
}

// ---------------------------------------------------------------------------
// Head: out[i,:5] = concat(xd[ld[i]], xp[lp[i]]) @ Wh[512,5] + bh
// One wave per label row; Wh staged in LDS.
// ---------------------------------------------------------------------------
__global__ __launch_bounds__(256)
void head_kernel(const float* __restrict__ xd, const float* __restrict__ xp,
                 const int* __restrict__ ld, const int* __restrict__ lp,
                 const float* __restrict__ hw, const float* __restrict__ hb,
                 float* __restrict__ out, int NLb) {
  __shared__ float sw[512 * 5];
  __shared__ float sb[5];
  for (int i = threadIdx.x; i < 512 * 5; i += 256) sw[i] = hw[i];
  if (threadIdx.x < 5) sb[threadIdx.x] = hb[threadIdx.x];
  __syncthreads();

  const int gw   = (blockIdx.x * blockDim.x + threadIdx.x) >> 5;
  const int lane = threadIdx.x & 31;
  if (gw >= NLb) return;

  float acc[5] = {0.f, 0.f, 0.f, 0.f, 0.f};
  const float* xr = xd + (size_t)ld[gw] * HID;
  for (int j = lane; j < HID; j += 32) {
    const float xv = xr[j];
#pragma unroll
    for (int o = 0; o < 5; ++o) acc[o] += xv * sw[j * 5 + o];
  }
  xr = xp + (size_t)lp[gw] * HID;
  for (int j = lane; j < HID; j += 32) {
    const float xv = xr[j];
#pragma unroll
    for (int o = 0; o < 5; ++o) acc[o] += xv * sw[(HID + j) * 5 + o];
  }
#pragma unroll
  for (int o = 0; o < 5; ++o) {
    float p = acc[o];
#pragma unroll
    for (int m = 16; m > 0; m >>= 1) p += __shfl_xor(p, m, 32);
    if (lane == 0) out[(size_t)gw * 5 + o] = p + sb[o];
  }
}

// ---------------------------------------------------------------------------
// Host orchestration
// ---------------------------------------------------------------------------
extern "C" void kernel_launch(void* const* d_in, const int* in_sizes, int n_in,
                              void* d_out, int out_size, void* d_ws, size_t ws_size,
                              hipStream_t stream) {
  (void)in_sizes; (void)out_size; (void)ws_size;

  const float* x_drug = (const float*)d_in[0];
  const float* x_prot = (const float*)d_in[1];
  const int* edp_s = (const int*)d_in[2];
  const int* edp_d = (const int*)d_in[3];
  const int* epd_s = (const int*)d_in[4];
  const int* epd_d = (const int*)d_in[5];
  const int* lbl_d = (const int*)d_in[6];
  const int* lbl_p = (const int*)d_in[7];

  // Params: leaves in setup_inputs()/make_params() insertion order.
  // Fallback: if harness packs params as one flat buffer, walk offsets instead.
  int pi = 8;
  const bool packed = (n_in <= 9);
  const float* P = packed ? (const float*)d_in[8] : nullptr;
  size_t po = 0;
  auto nxt = [&](size_t n) -> const float* {
    if (packed) { const float* p = P + po; po += n; return p; }
    return (const float*)d_in[pi++];
  };

  const float* in_drug_w = nxt(128 * HID);  const float* in_drug_b = nxt(HID);
  const float* in_prot_w = nxt(1280 * HID); const float* in_prot_b = nxt(HID);
  const float* head_w    = nxt(512 * 5);    const float* head_b    = nxt(5);

  struct Lp {
    const float *kdw,*kdb,*qdw,*qdb,*vdw,*vdb,*adw,*adb,*skd;
    const float *kpw,*kpb,*qpw,*qpb,*vpw,*vpb,*apw,*apb,*skp;
    const float *arel_dp,*mrel_dp,*prel_dp,*arel_pd,*mrel_pd,*prel_pd;
  } L[3];
  for (int l = 0; l < 3; ++l) {
    L[l].kdw = nxt(HID*HID); L[l].kdb = nxt(HID);
    L[l].qdw = nxt(HID*HID); L[l].qdb = nxt(HID);
    L[l].vdw = nxt(HID*HID); L[l].vdb = nxt(HID);
    L[l].adw = nxt(HID*HID); L[l].adb = nxt(HID);
    L[l].skd = nxt(1);
    L[l].kpw = nxt(HID*HID); L[l].kpb = nxt(HID);
    L[l].qpw = nxt(HID*HID); L[l].qpb = nxt(HID);
    L[l].vpw = nxt(HID*HID); L[l].vpb = nxt(HID);
    L[l].apw = nxt(HID*HID); L[l].apb = nxt(HID);
    L[l].skp = nxt(1);
    L[l].arel_dp = nxt(NHEAD*DHEAD*DHEAD);
    L[l].mrel_dp = nxt(NHEAD*DHEAD*DHEAD);
    L[l].prel_dp = nxt(NHEAD);
    L[l].arel_pd = nxt(NHEAD*DHEAD*DHEAD);
    L[l].mrel_pd = nxt(NHEAD*DHEAD*DHEAD);
    L[l].prel_pd = nxt(NHEAD);
  }

  // workspace carve (~280 MB of f32)
  float* wsf = (float*)d_ws;
  size_t off = 0;
  auto take = [&](size_t n) { float* p = wsf + off; off += n; return p; };
  float* xdb[2] = { take((size_t)ND * HID), take((size_t)ND * HID) };
  float* xpb[2] = { take((size_t)NP * HID), take((size_t)NP * HID) };
  float* kd = take((size_t)ND * HID); float* qd = take((size_t)ND * HID);
  float* vd = take((size_t)ND * HID);
  float* kp = take((size_t)NP * HID); float* qp = take((size_t)NP * HID);
  float* vp = take((size_t)NP * HID);
  float* msg_d = take((size_t)ND * HID);
  float* msg_p = take((size_t)NP * HID);
  float* alpha = take((size_t)NE * NHEAD);
  float* segmax = take((size_t)NP * NHEAD);
  float* segsum = take((size_t)NP * NHEAD);

  auto gemm = [&](const float* A, const float* Wg, const float* b, float* C,
                  int M, int K, int N, int gelu, const float* mixX, const float* sk) {
    dim3 grid(N / 64, (M + 63) / 64);
    wmma_gemm_bias<<<grid, dim3(256), 0, stream>>>(A, Wg, b, C, M, K, N, gelu, mixX, sk);
  };
  auto headwise = [&](float* X, const float* rel, int M) {  // in-place
    dim3 grid(NHEAD, (M + 63) / 64);
    wmma_headwise<<<grid, dim3(256), 0, stream>>>(X, rel, X, M);
  };
  auto fill = [&](float* p, float v, int n) {
    fill_kernel<<<(n + 255) / 256, 256, 0, stream>>>(p, v, n);
  };

  // input projections
  gemm(x_drug, in_drug_w, in_drug_b, xdb[0], ND, 128, HID, 0, nullptr, nullptr);
  gemm(x_prot, in_prot_w, in_prot_b, xpb[0], NP, 1280, HID, 0, nullptr, nullptr);

  int cur = 0;
  const int edgeWaveBlocks = (NE * 32 + 255) / 256;
  const int ehBlocks = (NE * NHEAD + 255) / 256;

  for (int l = 0; l < 3; ++l) {
    float* xd = xdb[cur];      float* xp = xpb[cur];
    float* xdn = xdb[cur ^ 1]; float* xpn = xpb[cur ^ 1];

    gemm(xd, L[l].kdw, L[l].kdb, kd, ND, HID, HID, 0, nullptr, nullptr);
    gemm(xd, L[l].qdw, L[l].qdb, qd, ND, HID, HID, 0, nullptr, nullptr);
    gemm(xd, L[l].vdw, L[l].vdb, vd, ND, HID, HID, 0, nullptr, nullptr);
    gemm(xp, L[l].kpw, L[l].kpb, kp, NP, HID, HID, 0, nullptr, nullptr);
    gemm(xp, L[l].qpw, L[l].qpb, qp, NP, HID, HID, 0, nullptr, nullptr);
    gemm(xp, L[l].vpw, L[l].vpb, vp, NP, HID, HID, 0, nullptr, nullptr);

    // relation transforms (in-place: kd->k_rel etc.)
    headwise(kd, L[l].arel_dp, ND);
    headwise(vd, L[l].mrel_dp, ND);
    headwise(kp, L[l].arel_pd, NP);
    headwise(vp, L[l].mrel_pd, NP);

    // dp relation: drug -> prot
    fill(segmax, -INFINITY, NP * NHEAD);
    fill(segsum, 0.0f, NP * NHEAD);
    fill(msg_p, 0.0f, NP * HID);
    edge_logits_kernel<<<edgeWaveBlocks, 256, 0, stream>>>(kd, qp, edp_s, edp_d,
                                                           L[l].prel_dp, alpha, NE);
    seg_max_kernel<<<ehBlocks, 256, 0, stream>>>(alpha, edp_d, segmax, NE);
    seg_exp_sum_kernel<<<ehBlocks, 256, 0, stream>>>(alpha, edp_d, segmax, segsum, NE);
    seg_aggregate_kernel<<<edgeWaveBlocks, 256, 0, stream>>>(alpha, segsum, vd,
                                                             edp_s, edp_d, msg_p, NE);

    // pd relation: prot -> drug
    fill(segmax, -INFINITY, ND * NHEAD);
    fill(segsum, 0.0f, ND * NHEAD);
    fill(msg_d, 0.0f, ND * HID);
    edge_logits_kernel<<<edgeWaveBlocks, 256, 0, stream>>>(kp, qd, epd_s, epd_d,
                                                           L[l].prel_pd, alpha, NE);
    seg_max_kernel<<<ehBlocks, 256, 0, stream>>>(alpha, epd_d, segmax, NE);
    seg_exp_sum_kernel<<<ehBlocks, 256, 0, stream>>>(alpha, epd_d, segmax, segsum, NE);
    seg_aggregate_kernel<<<edgeWaveBlocks, 256, 0, stream>>>(alpha, segsum, vp,
                                                             epd_s, epd_d, msg_d, NE);

    // update (gelu fused into A-stage; sigmoid-skip mix fused into epilogue)
    gemm(msg_d, L[l].adw, L[l].adb, xdn, ND, HID, HID, 1, xd, L[l].skd);
    gemm(msg_p, L[l].apw, L[l].apb, xpn, NP, HID, HID, 1, xp, L[l].skp);
    cur ^= 1;
  }

  head_kernel<<<(NLBL * 32 + 255) / 256, 256, 0, stream>>>(
      xdb[cur], xpb[cur], lbl_d, lbl_p, head_w, head_b, (float*)d_out, NLBL);
}
